// HierarchicalVQEncoder_79078937854122
// MI455X (gfx1250) — matrix operand order repaired and checked by reading, not compiled, not run on hardware
//
#include <hip/hip_runtime.h>
#include <hip/hip_bf16.h>

// ---- problem constants ----
#define N_ROWS 32768
#define D_INN  4096
#define D_HID  256
#define M_TILE 32
#define N_WGS  (N_ROWS / M_TILE)   // 1024
// LDS strides (padded for bank behavior / 16B alignment)
#define SRES   260                  // f32 stride for 256-wide row panels
#define SAB    264                  // bf16(half) stride for A staging (528B rows, 16B aligned)
#define SBB    40                   // bf16(half) stride for transposed B staging (80B rows)

typedef __attribute__((ext_vector_type(16))) __bf16        v16bf;
typedef __attribute__((ext_vector_type(8)))  float         v8f;
typedef __attribute__((ext_vector_type(4)))  unsigned int  u32x4;

union Frag { v16bf v; u32x4 q[2]; };

__device__ __forceinline__ unsigned short f2bf(float f) {
  union { float f; unsigned int u; } c; c.f = f;
  unsigned int u = c.u;
  u += 0x7FFFu + ((u >> 16) & 1u);       // round-to-nearest-even
  return (unsigned short)(u >> 16);
}

// A fragment: 16x32 bf16, lanes 0-15 row M=lane K=0..15, lanes 16-31 K=16..31
__device__ __forceinline__ void load_afrag(const unsigned short* sA, int mBase, int kOff,
                                           int lane, Frag& a) {
  int m = mBase + (lane & 15);
  int k = kOff + ((lane >> 4) << 4);
  const u32x4* p = (const u32x4*)(sA + m * SAB + k);
  a.q[0] = p[0];        // 8 halves (16B)
  a.q[1] = p[1];        // +8 halves (+16B)
}

// B fragment: 32x16 (KxN) bf16, staged transposed sB[n][k]; lanes 0-15 K=0..15, 16-31 K=16..31
__device__ __forceinline__ void load_bfrag(const unsigned short* sB, int nBase,
                                           int lane, Frag& b) {
  int n = nBase + (lane & 15);
  int k = (lane >> 4) << 4;
  const u32x4* p = (const u32x4*)(sB + n * SBB + k);
  b.q[0] = p[0];
  b.q[1] = p[1];
}

// C/D layout: element r -> M = (lane>>4)*8 + r, N = lane&15
__device__ __forceinline__ void store_panel_lds(float* dst, const v8f acc[4], const float* bias,
                                                int mBase, int nWave, int lane) {
#pragma unroll
  for (int nt = 0; nt < 4; ++nt) {
    int col = nWave + nt * 16 + (lane & 15);
    float b = bias[col];
#pragma unroll
    for (int r = 0; r < 8; ++r) {
      int m = mBase + ((lane >> 4) << 3) + r;
      dst[m * SRES + col] = acc[nt][r] + b;
    }
  }
  __syncthreads();
}

// GEMM over a pre-converted bf16 A panel (M_TILE x KTOT in sA) against W[k][n0+0..255]
template <int KTOT>
__device__ __forceinline__ void gemm_panel(const unsigned short* sA, unsigned short* sB,
                                           const float* __restrict__ W, int ldW, int n0,
                                           int t, int lane, int mBase, int nWave, v8f acc[4]) {
  for (int k0 = 0; k0 < KTOT; k0 += 32) {
#pragma unroll 8
    for (int kk = 0; kk < 32; ++kk)
      sB[t * SBB + kk] = f2bf(W[(size_t)(k0 + kk) * ldW + n0 + t]);
    __syncthreads();
    Frag a; load_afrag(sA, mBase, k0, lane, a);
#pragma unroll
    for (int nt = 0; nt < 4; ++nt) {
      Frag b; load_bfrag(sB, nWave + nt * 16, lane, b);
      acc[nt] = __builtin_amdgcn_wmma_f32_16x16x32_bf16(
          false, a.v, false, b.v, (short)0, acc[nt], false, false);
    }
    __syncthreads();
  }
}

// one VQ level: optional projection GEMM, nearest-codebook, qsum/residual update
template <int LVL, int KC>
__device__ __forceinline__ void vq_level(float* sRes, float* sQsum, float* sZp,
                                         unsigned short* sABf, unsigned short* sBbf,
                                         float* sDist, float* sCbn, int* sIdx,
                                         const float* __restrict__ pw, const float* __restrict__ pb,
                                         const float* __restrict__ cb,
                                         float* __restrict__ out, float& commitAcc,
                                         int row0, int t, int lane, int mBase, int nWave) {
  const float* zp;
  if constexpr (LVL == 0) {
    zp = sRes;                                   // identity projection
  } else {
    // residual -> bf16 A panel (full width, once per level)
#pragma unroll
    for (int i = 0; i < M_TILE; ++i) {
      int lin = i * 256 + t;
      int r = lin >> 8, c = lin & 255;
      sABf[r * SAB + c] = f2bf(sRes[r * SRES + c]);
    }
    __syncthreads();
    v8f pacc[4] = {};
    gemm_panel<D_HID>(sABf, sBbf, pw, D_HID, 0, t, lane, mBase, nWave, pacc);
    store_panel_lds(sZp, pacc, pb, mBase, nWave, lane);
    zp = sZp;
  }
  // codebook squared norms (tiny)
  if (t < KC) {
    float s = 0.f;
#pragma unroll 4
    for (int d = 0; d < 256; ++d) { float v = cb[t * 256 + d]; s = __builtin_fmaf(v, v, s); }
    sCbn[t] = s;
  }
  __syncthreads();
  // scores: ||cb_k||^2 - 2 * zp . cb_k   (||zp||^2 constant per row -> dropped for argmin)
  for (int p = t; p < M_TILE * KC; p += 256) {
    int r = p / KC, k = p % KC;
    const float* zr = zp + r * SRES;
    const float* ck = cb + k * 256;
    float dot = 0.f;
#pragma unroll 8
    for (int d = 0; d < 256; ++d) dot = __builtin_fmaf(zr[d], ck[d], dot);
    sDist[r * 32 + k] = sCbn[k] - 2.f * dot;
  }
  __syncthreads();
  if (t < M_TILE) {
    float bd = sDist[t * 32]; int bi = 0;
#pragma unroll
    for (int k = 1; k < KC; ++k) { float d = sDist[t * 32 + k]; if (d < bd) { bd = d; bi = k; } }
    sIdx[t] = bi;
    // indices tail of d_out: base = 32768*4096 + 6
    out[(size_t)134217734u + (size_t)LVL * N_ROWS + row0 + t] = (float)bi;
  }
  __syncthreads();
  // straight-through forward: qsum += q; commit += (q - res)^2; res -= q
  {
    const int c = t;
#pragma unroll 4
    for (int r = 0; r < M_TILE; ++r) {
      float q  = cb[sIdx[r] * 256 + c];
      float rr = sRes[r * SRES + c];
      float d  = q - rr;
      commitAcc = __builtin_fmaf(d, d, commitAcc);
      sQsum[r * SRES + c] += q;
      sRes[r * SRES + c]  = rr - q;
    }
  }
  __syncthreads();
}

__global__ __launch_bounds__(256) void vq_mega(
    const float* __restrict__ x,   const float* __restrict__ Win, const float* __restrict__ bin,
    const float* __restrict__ pw1, const float* __restrict__ pb1,
    const float* __restrict__ pw2, const float* __restrict__ pb2,
    const float* __restrict__ pw3, const float* __restrict__ pb3,
    const float* __restrict__ cb0, const float* __restrict__ cb1,
    const float* __restrict__ cb2, const float* __restrict__ cb3,
    const float* __restrict__ Wout, const float* __restrict__ bout,
    float* __restrict__ out, float* __restrict__ partials) {
  __shared__ float sRes[M_TILE * SRES];                      // residual (f32)
  __shared__ float sQsum[M_TILE * SRES];                     // quantized sum (f32)
  __shared__ float sZp[M_TILE * SRES];                       // projected residual (f32)
  __shared__ __align__(16) unsigned short sABf[M_TILE * SAB];// bf16 A staging
  __shared__ __align__(16) unsigned short sBbf[256 * SBB];   // bf16 B staging (transposed)
  __shared__ float sDist[M_TILE * 32];
  __shared__ float sCbn[32];
  __shared__ int   sIdx[M_TILE];
  __shared__ float sRed[256];

  const int t = threadIdx.x, lane = t & 31, wid = t >> 5;
  const int mBase = (wid & 1) * 16;        // 2 waves along M
  const int nWave = (wid >> 1) * 64;       // 4 waves along N (64 cols each)
  const int row0 = blockIdx.x * M_TILE;

  // ---------------- Phase 1: z = x @ W_in + b_in ----------------
  v8f zacc[4] = {};
  for (int k0 = 0; k0 < D_INN; k0 += 32) {
#pragma unroll
    for (int i = 0; i < 4; ++i) {          // A tile 32x32 f32 -> bf16
      int lin = i * 256 + t;
      int r = lin >> 5, kk = lin & 31;
      sABf[r * SAB + kk] = f2bf(x[(size_t)(row0 + r) * D_INN + k0 + kk]);
    }
#pragma unroll 8
    for (int kk = 0; kk < 32; ++kk)        // B tile 32x256 transposed
      sBbf[t * SBB + kk] = f2bf(Win[(size_t)(k0 + kk) * D_HID + t]);
    __syncthreads();
    Frag a; load_afrag(sABf, mBase, 0, lane, a);
#pragma unroll
    for (int nt = 0; nt < 4; ++nt) {
      Frag b; load_bfrag(sBbf, nWave + nt * 16, lane, b);
      zacc[nt] = __builtin_amdgcn_wmma_f32_16x16x32_bf16(
          false, a.v, false, b.v, (short)0, zacc[nt], false, false);
    }
    __syncthreads();
  }
  for (int i = t; i < M_TILE * SRES; i += 256) sQsum[i] = 0.f;
  store_panel_lds(sRes, zacc, bin, mBase, nWave, lane);      // has trailing barrier

  // ---------------- Phase 2: 4 VQ levels ----------------
  float c0 = 0.f, c1 = 0.f, c2 = 0.f, c3 = 0.f;
  vq_level<0, 32>(sRes, sQsum, sZp, sABf, sBbf, sDist, sCbn, sIdx,
                  nullptr, nullptr, cb0, out, c0, row0, t, lane, mBase, nWave);
  vq_level<1, 16>(sRes, sQsum, sZp, sABf, sBbf, sDist, sCbn, sIdx,
                  pw1, pb1, cb1, out, c1, row0, t, lane, mBase, nWave);
  vq_level<2, 8>(sRes, sQsum, sZp, sABf, sBbf, sDist, sCbn, sIdx,
                 pw2, pb2, cb2, out, c2, row0, t, lane, mBase, nWave);
  vq_level<3, 4>(sRes, sQsum, sZp, sABf, sBbf, sDist, sCbn, sIdx,
                 pw3, pb3, cb3, out, c3, row0, t, lane, mBase, nWave);

  // ---------------- Phase 3: x_recon = qsum @ W_out + b_out, fused recon loss ----------------
#pragma unroll
  for (int i = 0; i < M_TILE; ++i) {
    int lin = i * 256 + t;
    int r = lin >> 8, c = lin & 255;
    sABf[r * SAB + c] = f2bf(sQsum[r * SRES + c]);
  }
  __syncthreads();
  float reconSum = 0.f;
  for (int cc = 0; cc < 16; ++cc) {
    int n0 = cc * 256;
    v8f oacc[4] = {};
    gemm_panel<D_HID>(sABf, sBbf, Wout, D_INN, n0, t, lane, mBase, nWave, oacc);
#pragma unroll
    for (int nt = 0; nt < 4; ++nt) {
      int col = nWave + nt * 16 + (lane & 15);
      int gcol = n0 + col;
      float bo = bout[gcol];
#pragma unroll
      for (int r = 0; r < 8; ++r) {
        int m = mBase + ((lane >> 4) << 3) + r;
        size_t gi = (size_t)(row0 + m) * D_INN + gcol;
        float v = oacc[nt][r] + bo;
        out[gi] = v;
        float dd = v - x[gi];
        reconSum = __builtin_fmaf(dd, dd, reconSum);
      }
    }
  }

  // ---------------- deterministic per-WG partials: [recon, commit0..3] ----------------
  float vals[5] = { reconSum, c0, c1, c2, c3 };
#pragma unroll
  for (int j = 0; j < 5; ++j) {
    __syncthreads();
    sRed[t] = vals[j];
    __syncthreads();
#pragma unroll
    for (int s = 128; s >= 1; s >>= 1) {
      if (t < s) sRed[t] += sRed[t + s];
      __syncthreads();
    }
    if (t == 0) partials[blockIdx.x * 5 + j] = sRed[0];
  }
}

__global__ __launch_bounds__(256) void vq_finalize(const float* __restrict__ partials,
                                                   float* __restrict__ out) {
  __shared__ float s[5][256];
  int t = threadIdx.x;
  float a0 = 0, a1 = 0, a2 = 0, a3 = 0, a4 = 0;
  for (int g = t; g < N_WGS; g += 256) {
    const float* p = partials + (size_t)g * 5;
    a0 += p[0]; a1 += p[1]; a2 += p[2]; a3 += p[3]; a4 += p[4];
  }
  s[0][t] = a0; s[1][t] = a1; s[2][t] = a2; s[3][t] = a3; s[4][t] = a4;
  __syncthreads();
  for (int st = 128; st >= 1; st >>= 1) {
    if (t < st) {
#pragma unroll
      for (int j = 0; j < 5; ++j) s[j][t] += s[j][t + st];
    }
    __syncthreads();
  }
  if (t == 0) {
    const size_t O = (size_t)N_ROWS * D_INN;
    float recon = s[0][0] * (1.0f / ((float)N_ROWS * (float)D_INN));
    float cm0 = s[1][0] * (1.0f / ((float)N_ROWS * 256.f));
    float cm1 = s[2][0] * (1.0f / ((float)N_ROWS * 256.f));
    float cm2 = s[3][0] * (1.0f / ((float)N_ROWS * 256.f));
    float cm3 = s[4][0] * (1.0f / ((float)N_ROWS * 256.f));
    out[O + 0] = recon + 0.25f * (cm0 + cm1 + cm2 + cm3);  // total
    out[O + 1] = recon;                                    // recon_loss
    out[O + 2] = cm0; out[O + 3] = cm1; out[O + 4] = cm2; out[O + 5] = cm3;
  }
}

extern "C" void kernel_launch(void* const* d_in, const int* in_sizes, int n_in,
                              void* d_out, int out_size, void* d_ws, size_t ws_size,
                              hipStream_t stream) {
  const float* x    = (const float*)d_in[0];
  const float* Win  = (const float*)d_in[1];
  const float* bin  = (const float*)d_in[2];
  const float* pw1  = (const float*)d_in[3];
  const float* pb1  = (const float*)d_in[4];
  const float* pw2  = (const float*)d_in[5];
  const float* pb2  = (const float*)d_in[6];
  const float* pw3  = (const float*)d_in[7];
  const float* pb3  = (const float*)d_in[8];
  const float* cb0  = (const float*)d_in[9];
  const float* cb1  = (const float*)d_in[10];
  const float* cb2  = (const float*)d_in[11];
  const float* cb3  = (const float*)d_in[12];
  const float* Wout = (const float*)d_in[13];
  const float* bout = (const float*)d_in[14];
  float* out = (float*)d_out;
  float* partials = (float*)d_ws;   // N_WGS*5 floats = 20 KB

  vq_mega<<<dim3(N_WGS), dim3(256), 0, stream>>>(x, Win, bin, pw1, pb1, pw2, pb2, pw3, pb3,
                                                 cb0, cb1, cb2, cb3, Wout, bout, out, partials);
  vq_finalize<<<dim3(1), dim3(256), 0, stream>>>(partials, out);
}
